// FlowEstimator3D_83786222010962
// MI455X (gfx1250) — matrix operand order repaired and checked by reading, not compiled, not run on hardware
//
#include <hip/hip_runtime.h>

// ---------------------------------------------------------------------------
// FlowEstimator3D (CamLiFlow) for MI455X / gfx1250, wave32 + WMMA.
//
// v2: point_conv uses a 32-point tile, 512 threads (16 waves):
//   - each wave amortizes every B (weight) fragment over TWO wmma A-tiles
//     (halves L2 weight traffic vs 16-pt tiles),
//   - K-dim split across two wave groups ([0,544) / [544,1056)) with an LDS
//     f32 partial reduction, so all 16 waves run WMMA concurrently.
//   - 84 KB LDS via dynamic shared memory (gfx1250: 320 KB / WGP).
// ---------------------------------------------------------------------------

typedef _Float16 v16h __attribute__((ext_vector_type(16)));
typedef _Float16 v8h  __attribute__((ext_vector_type(8)));
typedef float    v8f  __attribute__((ext_vector_type(8)));

#define B_    4
#define N_    8192
#define K_    16
#define C_    128
#define CC_   131          // C + 3
#define KDIM  1048         // 8 * 131
#define KPAD  1056         // 33 * 32
#define KSPLIT 544         // 17 * 32 ; second half = 16 * 32
#define PTILE 32           // points per point_conv workgroup
#define TILE  16           // points per mlp workgroup
#define NEG   0.1f

// dynamic LDS layout for point_conv
#define AGG_BYTES  (PTILE * KPAD * 2)            // 67584
#define PART_FLOATS (8 * 2 * 32 * 8)             // 4096
#define PC_LDS_BYTES (AGG_BYTES + PART_FLOATS * 4)  // 83968

__device__ __forceinline__ float lrelu(float x) { return x > 0.f ? x : NEG * x; }

__device__ __forceinline__ v16h mk16(v8h lo, v8h hi) {
  return __builtin_shufflevector(lo, hi, 0,1,2,3,4,5,6,7,8,9,10,11,12,13,14,15);
}

// One wave, one 16x16 D tile: A row-major [16 x kdim] f16, B row-major
// (row = output col) f16, f32 accum. Pointers include per-lane hi offsets.
__device__ __forceinline__ v8f wmma_rowrow(const _Float16* aptr,
                                           const _Float16* bptr, int kdim) {
  v8f c = {};
  for (int kk = 0; kk < kdim; kk += 32) {
    v8h al = *reinterpret_cast<const v8h*>(aptr + kk);
    v8h ah = *reinterpret_cast<const v8h*>(aptr + kk + 16);
    v8h bl = *reinterpret_cast<const v8h*>(bptr + kk);
    v8h bh = *reinterpret_cast<const v8h*>(bptr + kk + 16);
    c = __builtin_amdgcn_wmma_f32_16x16x32_f16(false, mk16(al, ah), false,
                                               mk16(bl, bh), (short)0, c,
                                               false, false);
  }
  return c;
}

// ---- fp32 -> f16 bulk convert ---------------------------------------------
__global__ __launch_bounds__(256) void cvt_f16_kernel(const float* __restrict__ src,
                                                      _Float16* __restrict__ dst,
                                                      int n) {
  int i = blockIdx.x * 256 + threadIdx.x;
  if (i < n) dst[i] = (_Float16)src[i];
}

// ---- fp32 [rows,sc] -> f16 [rows,dc] with zero K-padding ------------------
__global__ __launch_bounds__(256) void padw_kernel(const float* __restrict__ src,
                                                   _Float16* __restrict__ dst,
                                                   int rows, int sc, int dc) {
  int i = blockIdx.x * 256 + threadIdx.x;
  if (i >= rows * dc) return;
  int r = i / dc, c = i - r * dc;
  dst[i] = (c < sc) ? (_Float16)src[r * sc + c] : (_Float16)0.f;
}

// ---- fused PointConv: gather+weightnet+agg (VALU) -> 1048->128 GEMM (WMMA) -
__global__ __launch_bounds__(512) void point_conv_kernel(
    const float*    __restrict__ xyz,   // [B,3,N] fp32
    const _Float16* __restrict__ feat,  // [B,128,N] f16
    const int*      __restrict__ idx,   // [B,N,K]
    const float*    __restrict__ wn_w,  // [8,3]
    const float*    __restrict__ wn_b,  // [8]
    const _Float16* __restrict__ lw,    // [128,KPAD] f16 (row o = output chan)
    const float*    __restrict__ lb,    // [128]
    _Float16*       __restrict__ out)   // [B,128,N] f16
{
  extern __shared__ char smem[];
  _Float16* AGG  = reinterpret_cast<_Float16*>(smem);          // [32][KPAD]
  float*    PART = reinterpret_cast<float*>(smem + AGG_BYTES); // [8][2][32][8]

  const int tid = threadIdx.x;            // 0..511
  const int blk = blockIdx.x;
  const int b   = blk / (N_ / PTILE);
  const int n0  = (blk % (N_ / PTILE)) * PTILE;

  // zero the K padding [1048,1056) of every agg row
  if (tid < 256) {
    int p = tid >> 3, t = tid & 7;
    AGG[p * KPAD + KDIM + t] = (_Float16)0.f;
  }

  // weight-net params (block-uniform -> scalar regs)
  float ww[24], wb[8];
  #pragma unroll
  for (int i = 0; i < 24; ++i) ww[i] = wn_w[i];
  #pragma unroll
  for (int i = 0; i < 8; ++i)  wb[i] = wn_b[i];

  // ------------------ stage 1: agg[p][o*131+c], 16 lanes per point ----------
  {
    const int p  = tid >> 4;   // 0..31
    const int lc = tid & 15;   // 0..15
    const int n  = n0 + p;
    const float cx = xyz[(b * 3 + 0) * N_ + n];
    const float cy = xyz[(b * 3 + 1) * N_ + n];
    const float cz = xyz[(b * 3 + 2) * N_ + n];
    const int nc = (lc < 3) ? 9 : 8;   // channels c = lc + 16*i, c < 131

    float acc[72];
    #pragma unroll
    for (int i = 0; i < 72; ++i) acc[i] = 0.f;

    for (int k = 0; k < K_; ++k) {
      const int j = idx[(b * N_ + n) * K_ + k];
      const float rx = xyz[(b * 3 + 0) * N_ + j] - cx;
      const float ry = xyz[(b * 3 + 1) * N_ + j] - cy;
      const float rz = xyz[(b * 3 + 2) * N_ + j] - cz;
      float w[8];
      #pragma unroll
      for (int o = 0; o < 8; ++o)
        w[o] = lrelu(ww[o * 3 + 0] * rx + ww[o * 3 + 1] * ry +
                     ww[o * 3 + 2] * rz + wb[o]);
      #pragma unroll
      for (int i = 0; i < 9; ++i) {
        if (i < nc) {
          const int c = lc + 16 * i;
          const float f = (c < 3)
              ? xyz[(b * 3 + c) * N_ + j]
              : (float)feat[((size_t)b * C_ + (c - 3)) * N_ + j];
          #pragma unroll
          for (int o = 0; o < 8; ++o) acc[i * 8 + o] += w[o] * f;
        }
      }
    }
    #pragma unroll
    for (int i = 0; i < 9; ++i) {
      if (i < nc) {
        const int c = lc + 16 * i;
        #pragma unroll
        for (int o = 0; o < 8; ++o)
          AGG[p * KPAD + o * CC_ + c] = (_Float16)acc[i * 8 + o];
      }
    }
  }
  __syncthreads();

  // ---- stage 2: split-K WMMA GEMM, 16 waves = 8 col-slabs x 2 K-halves -----
  {
    const int wave = tid >> 5;         // 0..15
    const int lane = tid & 31;
    const int ol = lane & 15, hi = lane >> 4;
    const int os = wave & 7;           // output col slab
    const int kh = wave >> 3;          // K half
    const int o0 = os * 16;
    const int kbeg = kh ? KSPLIT : 0;
    const int kend = kh ? KPAD : KSPLIT;

    v8f c0 = {}, c1 = {};
    const _Float16* bp  = lw + (size_t)(o0 + ol) * KPAD + hi * 8;
    const _Float16* ap0 = AGG + (lane & 15) * KPAD + hi * 8;
    const _Float16* ap1 = AGG + (16 + (lane & 15)) * KPAD + hi * 8;
    for (int kk = kbeg; kk < kend; kk += 32) {
      v8h bl = *reinterpret_cast<const v8h*>(bp + kk);
      v8h bh = *reinterpret_cast<const v8h*>(bp + kk + 16);
      v16h bmat = mk16(bl, bh);
      v8h a0l = *reinterpret_cast<const v8h*>(ap0 + kk);
      v8h a0h = *reinterpret_cast<const v8h*>(ap0 + kk + 16);
      v8h a1l = *reinterpret_cast<const v8h*>(ap1 + kk);
      v8h a1h = *reinterpret_cast<const v8h*>(ap1 + kk + 16);
      c0 = __builtin_amdgcn_wmma_f32_16x16x32_f16(false, mk16(a0l, a0h), false,
                                                  bmat, (short)0, c0, false, false);
      c1 = __builtin_amdgcn_wmma_f32_16x16x32_f16(false, mk16(a1l, a1h), false,
                                                  bmat, (short)0, c1, false, false);
    }

    if (kh == 1) {  // publish partials
      *reinterpret_cast<v8f*>(PART + ((os * 2 + 0) * 32 + lane) * 8) = c0;
      *reinterpret_cast<v8f*>(PART + ((os * 2 + 1) * 32 + lane) * 8) = c1;
    }
    __syncthreads();
    if (kh == 0) {  // reduce + epilogue
      c0 += *reinterpret_cast<const v8f*>(PART + ((os * 2 + 0) * 32 + lane) * 8);
      c1 += *reinterpret_cast<const v8f*>(PART + ((os * 2 + 1) * 32 + lane) * 8);
      const float bias = lb[o0 + ol];
      v8h r0, r1;
      #pragma unroll
      for (int r = 0; r < 8; ++r) {
        r0[r] = (_Float16)lrelu(c0[r] + bias);
        r1[r] = (_Float16)lrelu(c1[r] + bias);
      }
      // D layout: lane holds col o0+ol, points n0+hi*8 .. +7 (consecutive)
      _Float16* op = out + ((size_t)b * C_ + (o0 + ol)) * N_ + n0 + hi * 8;
      *reinterpret_cast<v8h*>(op)      = r0;   // points n0    .. n0+15
      *reinterpret_cast<v8h*>(op + 16) = r1;   // points n0+16 .. n0+31
    }
  }
}

// ---- fused MLP: 128->128 -> 128->64 (flow_feat) -> 64->3 (flow) -----------
__global__ __launch_bounds__(256) void mlp_kernel(
    const _Float16* __restrict__ f2,  // [B,128,N] f16
    const _Float16* __restrict__ w1,  // [128,128] f16
    const float*    __restrict__ b1,  // [128]
    const _Float16* __restrict__ w2,  // [64,128] f16
    const float*    __restrict__ b2,  // [64]
    const float*    __restrict__ cw,  // [3,64] fp32
    const float*    __restrict__ cb,  // [3]
    float*          __restrict__ ff,  // [B,64,N] flow_feat (fp32 out)
    float*          __restrict__ fl)  // [B,3,N]  flow (fp32 out)
{
  __shared__ _Float16 A1[TILE * 128];
  __shared__ _Float16 H1[TILE * 128];
  __shared__ _Float16 H2[TILE * 64];

  const int tid = threadIdx.x;
  const int blk = blockIdx.x;
  const int b   = blk / (N_ / TILE);
  const int n0  = (blk % (N_ / TILE)) * TILE;

  // load input tile [p][c] row-major
  {
    const int p  = tid >> 4;
    const int c0 = (tid & 15) * 8;
    const int n  = n0 + p;
    #pragma unroll
    for (int t = 0; t < 8; ++t)
      A1[p * 128 + c0 + t] = f2[((size_t)b * C_ + c0 + t) * N_ + n];
  }
  __syncthreads();

  const int wave = tid >> 5, lane = tid & 31;
  const int ol = lane & 15, hi = lane >> 4;
  const int o0 = wave * 16;

  // GEMM1: 128 -> 128, LeakyReLU, into H1
  {
    v8f c = wmma_rowrow(&A1[(lane & 15) * 128 + hi * 8],
                        w1 + (o0 + ol) * 128 + hi * 8, 128);
    const float bias = b1[o0 + ol];
    #pragma unroll
    for (int r = 0; r < 8; ++r)
      H1[(r + hi * 8) * 128 + o0 + ol] = (_Float16)lrelu(c[r] + bias);
  }
  __syncthreads();

  // GEMM2: 128 -> 64, LeakyReLU -> flow_feat (global) + H2 (LDS). Waves 0..3.
  if (wave < 4) {
    v8f c = wmma_rowrow(&H1[(lane & 15) * 128 + hi * 8],
                        w2 + (o0 + ol) * 128 + hi * 8, 128);
    const float bias = b2[o0 + ol];
    float vals[8];
    #pragma unroll
    for (int r = 0; r < 8; ++r) vals[r] = lrelu(c[r] + bias);
    #pragma unroll
    for (int r = 0; r < 8; ++r)
      H2[(r + hi * 8) * 64 + o0 + ol] = (_Float16)vals[r];
    float* op = ff + ((size_t)b * 64 + o0 + ol) * N_ + n0 + hi * 8;
    #pragma unroll
    for (int r = 0; r < 8; ++r) op[r] = vals[r];
  }
  __syncthreads();

  // conv_last: 64 -> 3 (no activation), one lane per point
  if (tid < TILE) {
    const int p = tid, n = n0 + p;
    float s0 = cb[0], s1 = cb[1], s2 = cb[2];
    #pragma unroll
    for (int c = 0; c < 64; ++c) {
      const float h = (float)H2[p * 64 + c];
      s0 += cw[0 * 64 + c] * h;
      s1 += cw[1 * 64 + c] * h;
      s2 += cw[2 * 64 + c] * h;
    }
    fl[((size_t)b * 3 + 0) * N_ + n] = s0;
    fl[((size_t)b * 3 + 1) * N_ + n] = s1;
    fl[((size_t)b * 3 + 2) * N_ + n] = s2;
  }
}

// ---------------------------------------------------------------------------
extern "C" void kernel_launch(void* const* d_in, const int* in_sizes, int n_in,
                              void* d_out, int out_size, void* d_ws, size_t ws_size,
                              hipStream_t stream) {
  const float* xyz    = (const float*)d_in[0];
  const float* feat   = (const float*)d_in[1];
  const int*   idx    = (const int*)  d_in[2];
  const float* wn1_w  = (const float*)d_in[3];
  const float* wn1_b  = (const float*)d_in[4];
  const float* lin1_w = (const float*)d_in[5];
  const float* lin1_b = (const float*)d_in[6];
  const float* wn2_w  = (const float*)d_in[7];
  const float* wn2_b  = (const float*)d_in[8];
  const float* lin2_w = (const float*)d_in[9];
  const float* lin2_b = (const float*)d_in[10];
  const float* m1_w   = (const float*)d_in[11];
  const float* m1_b   = (const float*)d_in[12];
  const float* m2_w   = (const float*)d_in[13];
  const float* m2_b   = (const float*)d_in[14];
  const float* cl_w   = (const float*)d_in[15];
  const float* cl_b   = (const float*)d_in[16];

  // workspace carve (f16 intermediates + f16 padded weights)
  const size_t SZF = (size_t)B_ * C_ * N_ * sizeof(_Float16);      // 8 MB
  char* ws = (char*)d_ws;
  _Float16* f0 = (_Float16*)(ws);                                   // feat f16
  _Float16* f1 = (_Float16*)(ws + SZF);                             // pc1 out
  _Float16* f2 = (_Float16*)(ws + 2 * SZF);                         // pc2 out
  size_t off = 3 * SZF;
  _Float16* w1 = (_Float16*)(ws + off); off += (size_t)128 * KPAD * 2;
  _Float16* w2 = (_Float16*)(ws + off); off += (size_t)128 * KPAD * 2;
  _Float16* wm1 = (_Float16*)(ws + off); off += (size_t)128 * 128 * 2;
  _Float16* wm2 = (_Float16*)(ws + off); off += (size_t)64 * 128 * 2;

  float* out_ff = (float*)d_out;                       // [B,64,N]
  float* out_fl = out_ff + (size_t)B_ * 64 * N_;       // [B,3,N]

  // 1) precision staging
  {
    int n = B_ * C_ * N_;
    cvt_f16_kernel<<<(n + 255) / 256, 256, 0, stream>>>(feat, f0, n);
    int e1 = 128 * KPAD;
    padw_kernel<<<(e1 + 255) / 256, 256, 0, stream>>>(lin1_w, w1, 128, KDIM, KPAD);
    padw_kernel<<<(e1 + 255) / 256, 256, 0, stream>>>(lin2_w, w2, 128, KDIM, KPAD);
    int e2 = 128 * 128;
    padw_kernel<<<(e2 + 255) / 256, 256, 0, stream>>>(m1_w, wm1, 128, 128, 128);
    int e3 = 64 * 128;
    padw_kernel<<<(e3 + 255) / 256, 256, 0, stream>>>(m2_w, wm2, 64, 128, 128);
  }

  // 2) point_conv1 / point_conv2 (fused gather + split-K WMMA GEMM)
  const int pc_blocks = B_ * N_ / PTILE;   // 1024
  point_conv_kernel<<<pc_blocks, 512, PC_LDS_BYTES, stream>>>(
      xyz, f0, idx, wn1_w, wn1_b, w1, lin1_b, f1);
  point_conv_kernel<<<pc_blocks, 512, PC_LDS_BYTES, stream>>>(
      xyz, f1, idx, wn2_w, wn2_b, w2, lin2_b, f2);

  // 3) fused MLP + conv_last
  const int mlp_blocks = B_ * N_ / TILE;   // 2048
  mlp_kernel<<<mlp_blocks, 256, 0, stream>>>(f2, wm1, m1_b, wm2, m2_b,
                                             cl_w, cl_b, out_ff, out_fl);

  (void)in_sizes; (void)n_in; (void)out_size; (void)ws_size;
}
